// DeepSeekV3Attention_43619687858991
// MI455X (gfx1250) — matrix operand order repaired and checked by reading, not compiled, not run on hardware
//
#include <hip/hip_runtime.h>

typedef __attribute__((ext_vector_type(16))) __bf16 v16bf;
typedef __attribute__((ext_vector_type(8)))  float v8f;

#define B_ 4
#define S_ 2048
#define D_ 384
#define H_ 6
#define KVH_ 2
#define HD_ 64
#define GROUPS_ 3
#define M_ (B_ * S_)          // 8192 token rows

// ---------- CDNA5 async-to-LDS path (guarded; falls back to uint4 copy) ----------
#if __has_builtin(__builtin_amdgcn_global_load_async_to_lds_b128) && \
    __has_builtin(__builtin_amdgcn_s_wait_asynccnt)
#define HAVE_ASYNC 1
#else
#define HAVE_ASYNC 0
#endif

#if HAVE_ASYNC
typedef int v4i_a __attribute__((vector_size(16)));
typedef __attribute__((address_space(1))) v4i_a* gv4p_t;   // global
typedef __attribute__((address_space(3))) v4i_a* lv4p_t;   // LDS
#endif

// 16-byte aligned global -> LDS copy of 16 bytes
static __device__ __forceinline__ void async_b128(const unsigned short* g, unsigned short* l) {
#if HAVE_ASYNC
  __builtin_amdgcn_global_load_async_to_lds_b128(
      (gv4p_t)(void*)const_cast<unsigned short*>(g),
      (lv4p_t)(void*)l, 0, 0);
#else
  *(uint4*)l = *(const uint4*)g;
#endif
}

static __device__ __forceinline__ void async_wait() {
#if HAVE_ASYNC
  __builtin_amdgcn_s_wait_asynccnt(0);
#endif
}

// ---------- helpers ----------

static __device__ __forceinline__ unsigned short f2bf(float f) {
  union { float f; unsigned u; } c; c.f = f;
  unsigned r = c.u + 0x7FFFu + ((c.u >> 16) & 1u);   // round-to-nearest-even
  return (unsigned short)(r >> 16);
}

static __device__ __forceinline__ float fast_exp2(float x) {
#if __has_builtin(__builtin_amdgcn_exp2f)
  return __builtin_amdgcn_exp2f(x);
#else
  return exp2f(x);
#endif
}

union Frag { v16bf v; unsigned u[8]; };

// A fragment: 16x32 bf16 tile stored row-major [m][k] (k contiguous), row stride ld.
// ISA 7.12.2: lane m = L&15, half = L>>4; VGPR v holds K = (v/4)*16 + half*8 + (v%4)*2, +1
static __device__ __forceinline__ v16bf load_a_frag(const unsigned short* base, int ld, int lane) {
  int m = lane & 15, half = lane >> 4;
  Frag f;
  const unsigned short* row = base + m * ld;
#pragma unroll
  for (int v = 0; v < 8; ++v) {
    int k = ((v >> 2) << 4) + (half << 3) + ((v & 3) << 1);
    f.u[v] = *(const unsigned*)(row + k);
  }
  return f.v;
}

// B fragment: 32x16 bf16 matrix (K x N), sourced from a TRANSPOSED tile [n][k]
// (k contiguous, row stride ld). Lane n = L&15, half = L>>4; VGPR v holds K = (v/4)*16 + half*8 + (v%4)*2, +1
static __device__ __forceinline__ v16bf load_bt_frag(const unsigned short* base, int ld, int lane) {
  int n = lane & 15, half = lane >> 4;
  Frag f;
  const unsigned short* row = base + n * ld;
#pragma unroll
  for (int v = 0; v < 8; ++v) {
    int k = ((v >> 2) << 4) + (half << 3) + ((v & 3) << 1);
    f.u[v] = *(const unsigned*)(row + k);
  }
  return f.v;
}

#define WMMA_BF16(A, Bv, C) \
  __builtin_amdgcn_wmma_f32_16x16x32_bf16(false, (A), false, (Bv), (short)0, (C), false, false)

// ---------- fp32 -> bf16 cast ----------

__global__ void cast_f32_bf16(const float* __restrict__ src, unsigned short* __restrict__ dst, int n) {
  int i = blockIdx.x * blockDim.x + threadIdx.x;
  if (i < n) dst[i] = f2bf(src[i]);
}

// fp32 [K][N] -> bf16 [N][K] (weight pre-transpose; tiny, runs once)
__global__ void cast_transpose_bf16(const float* __restrict__ src, unsigned short* __restrict__ dst,
                                    int K, int N) {
  int i = blockIdx.x * blockDim.x + threadIdx.x;
  if (i < K * N) {
    int n = i / K, k = i % K;
    dst[i] = f2bf(src[(size_t)k * N + n]);
  }
}

// ---------- bf16 WMMA GEMM: C[M,N] = A[M,K] * W[K,N], W given pre-transposed as Wt[N][K] ----------
// grid = (M/64, N/64), block = 128 (4 waves). K multiple of 32, N multiple of 64.

__global__ __launch_bounds__(128) void gemm_bf16_f32(const unsigned short* __restrict__ A,
                                                     const unsigned short* __restrict__ Wt,
                                                     float* __restrict__ C, int N, int K) {
  __shared__ __align__(16) unsigned short at[64 * 32];   // [m][k]
  __shared__ __align__(16) unsigned short bt[64 * 32];   // [n][k]
  const int t = threadIdx.x, wave = t >> 5, lane = t & 31;
  const int gm0 = blockIdx.x * 64, gn0 = blockIdx.y * 64;

  v8f acc[4] = {};

  for (int kc = 0; kc < K; kc += 32) {
    // A tile: 64 rows x 32 elems = 256 x 16B chunks, 2 per thread (async -> LDS)
#pragma unroll
    for (int i = 0; i < 2; ++i) {
      int j = t + (i << 7);
      int m = j >> 2, c = (j & 3) << 3;
      async_b128(A + (size_t)(gm0 + m) * K + kc + c, at + m * 32 + c);
    }
    // Wt tile: straight copy thanks to pre-transposed weights
#pragma unroll
    for (int i = 0; i < 2; ++i) {
      int j = t + (i << 7);
      int n = j >> 2, c = (j & 3) << 3;
      async_b128(Wt + (size_t)(gn0 + n) * K + kc + c, bt + n * 32 + c);
    }
    async_wait();
    __syncthreads();

    v16bf af = load_a_frag(at + (wave << 4) * 32, 32, lane);
#pragma unroll
    for (int nc = 0; nc < 4; ++nc) {
      v16bf bf = load_bt_frag(bt + (nc << 4) * 32, 32, lane);
      acc[nc] = WMMA_BF16(af, bf, acc[nc]);
    }
    __syncthreads();
  }

  const int n = lane & 15, half = lane >> 4;
#pragma unroll
  for (int nc = 0; nc < 4; ++nc)
#pragma unroll
    for (int r = 0; r < 8; ++r) {
      int m = gm0 + (wave << 4) + r + (half << 3);
      C[(size_t)m * N + gn0 + (nc << 4) + n] = acc[nc][r];
    }
}

// ---------- RoPE + reshape to head-major bf16 ----------
// qp: [B*S, 384] f32, kp/vp: [B*S, 128] f32
// qo: [B,H,S,64] bf16, ko/vo: [B,KVH,S,64] bf16

__global__ __launch_bounds__(128) void rope_kernel(const float* __restrict__ qp,
                                                   const float* __restrict__ kp,
                                                   const float* __restrict__ vp,
                                                   const float* __restrict__ fc,
                                                   const float* __restrict__ fs,
                                                   unsigned short* __restrict__ qo,
                                                   unsigned short* __restrict__ ko,
                                                   unsigned short* __restrict__ vo) {
  const int bs = blockIdx.x;
  const int b = bs / S_, s = bs % S_;
  const int t = threadIdx.x;

  for (int idx = t; idx < 256; idx += 128) {
    if (idx < 192) {                 // q: 6 heads * 32 pairs
      int h = idx >> 5, d2 = idx & 31;
      const float* src = qp + (size_t)bs * (H_ * HD_) + h * HD_ + (d2 << 1);
      float x0 = src[0], x1 = src[1];
      float c = fc[s * 32 + d2], sn = fs[s * 32 + d2];
      size_t dst = (((size_t)(b * H_ + h)) * S_ + s) * HD_ + (d2 << 1);
      qo[dst]     = f2bf(x0 * c - x1 * sn);
      qo[dst + 1] = f2bf(x0 * sn + x1 * c);
    } else {                         // k: 2 heads * 32 pairs
      int j = idx - 192;
      int kvh = j >> 5, d2 = j & 31;
      const float* src = kp + (size_t)bs * (KVH_ * HD_) + kvh * HD_ + (d2 << 1);
      float x0 = src[0], x1 = src[1];
      float c = fc[s * 32 + d2], sn = fs[s * 32 + d2];
      size_t dst = (((size_t)(b * KVH_ + kvh)) * S_ + s) * HD_ + (d2 << 1);
      ko[dst]     = f2bf(x0 * c - x1 * sn);
      ko[dst + 1] = f2bf(x0 * sn + x1 * c);
    }
  }
  // v: 128 elems, no rope
  {
    int kvh = t >> 6, d = t & 63;
    vo[(((size_t)(b * KVH_ + kvh)) * S_ + s) * HD_ + d] =
        f2bf(vp[(size_t)bs * (KVH_ * HD_) + t]);
  }
}

// ---------- flash attention (non-causal, GQA), 64-key chunks ----------
// grid = B*H*(S/64), block = 128 (4 waves, each wave = one 16-query tile)
// O: [B, S, H*64] bf16

__global__ __launch_bounds__(128) void attn_kernel(const unsigned short* __restrict__ Q,
                                                   const unsigned short* __restrict__ Kb,
                                                   const unsigned short* __restrict__ V,
                                                   unsigned short* __restrict__ O) {
  __shared__ __align__(16) unsigned short kt[64 * 64];    // [key][d]   (B-frag for Q*K^T)
  __shared__ __align__(16) unsigned short vt[64 * 64];    // [d][key]   (B-frag for P*V)
  __shared__ __align__(16) unsigned short pt[4][16 * 64]; // per-wave P staging [m][k]

  const int t = threadIdx.x, wave = t >> 5, lane = t & 31;
  const int nqc = S_ / 64;
  const int bid = blockIdx.x;
  const int qc = bid % nqc;
  const int h  = (bid / nqc) % H_;
  const int b  = bid / (nqc * H_);
  const int kvh = h / GROUPS_;

  const unsigned short* qptr = Q  + (size_t)(b * H_ + h)     * S_ * HD_;
  const unsigned short* kptr = Kb + (size_t)(b * KVH_ + kvh) * S_ * HD_;
  const unsigned short* vptr = V  + (size_t)(b * KVH_ + kvh) * S_ * HD_;

  const int q0 = qc * 64 + wave * 16;
  // Q tile lives in registers as two A fragments (d 0..31, d 32..63)
  v16bf qf0 = load_a_frag(qptr + (size_t)q0 * HD_,      HD_, lane);
  v16bf qf1 = load_a_frag(qptr + (size_t)q0 * HD_ + 32, HD_, lane);

  float mrow[8], lrow[8];
  v8f acc[4] = {};
#pragma unroll
  for (int r = 0; r < 8; ++r) { mrow[r] = -__builtin_inff(); lrow[r] = 0.f; }

  const float sc = 0.125f * 1.44269504089f;   // 1/sqrt(64) * log2(e)

  for (int kc = 0; kc < S_; kc += 64) {
    // stage K chunk [64 keys][64 d]: straight copy, async -> LDS (512 x 16B, 4/thread)
#pragma unroll
    for (int i = 0; i < 4; ++i) {
      int j = t + (i << 7);
      int key = j >> 3, c = (j & 7) << 3;
      async_b128(kptr + (size_t)(kc + key) * HD_ + c, kt + key * HD_ + c);
    }
    // stage V chunk transposed to [d][key]
    {
      int key = t & 63, dh = (t >> 6) << 5;
#pragma unroll
      for (int i2 = 0; i2 < 32; ++i2) {
        int d = dh + i2;
        vt[d * 64 + key] = vptr[(size_t)(kc + key) * HD_ + d];
      }
    }
    // prefetch next chunk while this one computes (global_prefetch_b8)
    if (kc + 64 < S_) {
      __builtin_prefetch(kptr + (size_t)(kc + 64) * HD_ + t * 32, 0, 1);
      __builtin_prefetch(vptr + (size_t)(kc + 64) * HD_ + t * 32, 0, 1);
    }
    async_wait();
    __syncthreads();

    // scores: four 16x16 tiles (keys kc+16j), K-dim = HD = 64 -> 2 WMMAs each
    v8f s[4];
#pragma unroll
    for (int j = 0; j < 4; ++j) {
      v8f sj = {};
      v16bf b0 = load_bt_frag(kt + j * 16 * 64,      64, lane);
      sj = WMMA_BF16(qf0, b0, sj);
      v16bf b1 = load_bt_frag(kt + j * 16 * 64 + 32, 64, lane);
      sj = WMMA_BF16(qf1, b1, sj);
      s[j] = sj;
    }

    // online softmax, log2 domain; C layout: row m = r + 8*half, n = lane&15
    float alpha[8];
#pragma unroll
    for (int r = 0; r < 8; ++r) {
      float x0 = s[0][r] * sc, x1 = s[1][r] * sc, x2 = s[2][r] * sc, x3 = s[3][r] * sc;
      float cm = fmaxf(fmaxf(x0, x1), fmaxf(x2, x3));
      cm = fmaxf(cm, __shfl_xor(cm, 8, 16));
      cm = fmaxf(cm, __shfl_xor(cm, 4, 16));
      cm = fmaxf(cm, __shfl_xor(cm, 2, 16));
      cm = fmaxf(cm, __shfl_xor(cm, 1, 16));
      float mn = fmaxf(mrow[r], cm);
      alpha[r] = fast_exp2(mrow[r] - mn);
      mrow[r] = mn;
      float e0 = fast_exp2(x0 - mn);
      float e1 = fast_exp2(x1 - mn);
      float e2 = fast_exp2(x2 - mn);
      float e3 = fast_exp2(x3 - mn);
      float ps = (e0 + e1) + (e2 + e3);
      ps += __shfl_xor(ps, 8, 16);
      ps += __shfl_xor(ps, 4, 16);
      ps += __shfl_xor(ps, 2, 16);
      ps += __shfl_xor(ps, 1, 16);
      lrow[r] = lrow[r] * alpha[r] + ps;
      int m = r + ((lane >> 4) << 3), nn = lane & 15;
      unsigned short* pr = &pt[wave][m * 64];
      pr[nn]      = f2bf(e0);
      pr[16 + nn] = f2bf(e1);
      pr[32 + nn] = f2bf(e2);
      pr[48 + nn] = f2bf(e3);
    }
#pragma unroll
    for (int nc = 0; nc < 4; ++nc)
#pragma unroll
      for (int r = 0; r < 8; ++r) acc[nc][r] = acc[nc][r] * alpha[r];

    // P (16x64) * V (64x64): 2 A-frags x 4 d-chunks = 8 WMMAs
    v16bf pf0 = load_a_frag(pt[wave],      64, lane);
    v16bf pf1 = load_a_frag(pt[wave] + 32, 64, lane);
#pragma unroll
    for (int nc = 0; nc < 4; ++nc) {
      v16bf bv0 = load_bt_frag(vt + (nc << 4) * 64,      64, lane);
      acc[nc] = WMMA_BF16(pf0, bv0, acc[nc]);
      v16bf bv1 = load_bt_frag(vt + (nc << 4) * 64 + 32, 64, lane);
      acc[nc] = WMMA_BF16(pf1, bv1, acc[nc]);
    }
    __syncthreads();
  }

  // normalize and write O[b][q][h*64 + d] in bf16
  const int n = lane & 15, half = lane >> 4;
#pragma unroll
  for (int r = 0; r < 8; ++r) {
    float inv = 1.f / lrow[r];
    int m = q0 + r + (half << 3);
    size_t rowoff = ((size_t)b * S_ + m) * (size_t)(H_ * HD_) + h * HD_;
#pragma unroll
    for (int nc = 0; nc < 4; ++nc)
      O[rowoff + (nc << 4) + n] = f2bf(acc[nc][r] * inv);
  }
}

// ---------- host side ----------

static inline size_t align256(size_t x) { return (x + 255) & ~(size_t)255; }

extern "C" void kernel_launch(void* const* d_in, const int* in_sizes, int n_in,
                              void* d_out, int out_size, void* d_ws, size_t ws_size,
                              hipStream_t stream) {
  const float* hidden = (const float*)d_in[0];
  const float* Wq = (const float*)d_in[1];
  const float* Wk = (const float*)d_in[2];
  const float* Wv = (const float*)d_in[3];
  const float* Wo = (const float*)d_in[4];
  const float* fc = (const float*)d_in[5];
  const float* fs = (const float*)d_in[6];

  char* ws = (char*)d_ws;
  size_t off = 0;
  unsigned short* hid_bf = (unsigned short*)(ws + off); off += align256((size_t)M_ * D_ * 2);
  unsigned short* wqt = (unsigned short*)(ws + off); off += align256((size_t)D_ * H_ * HD_ * 2);
  unsigned short* wkt = (unsigned short*)(ws + off); off += align256((size_t)D_ * KVH_ * HD_ * 2);
  unsigned short* wvt = (unsigned short*)(ws + off); off += align256((size_t)D_ * KVH_ * HD_ * 2);
  unsigned short* wot = (unsigned short*)(ws + off); off += align256((size_t)H_ * HD_ * D_ * 2);
  float* qproj = (float*)(ws + off);                 size_t qproj_off = off;
                                                     off += align256((size_t)M_ * H_ * HD_ * 4);
  float* kproj = (float*)(ws + off);                 off += align256((size_t)M_ * KVH_ * HD_ * 4);
  float* vproj = (float*)(ws + off);                 off += align256((size_t)M_ * KVH_ * HD_ * 4);
  unsigned short* q_bf = (unsigned short*)(ws + off); off += align256((size_t)M_ * H_ * HD_ * 2);
  unsigned short* k_bf = (unsigned short*)(ws + off); off += align256((size_t)M_ * KVH_ * HD_ * 2);
  unsigned short* v_bf = (unsigned short*)(ws + off); off += align256((size_t)M_ * KVH_ * HD_ * 2);
  // attention output aliases the (by then dead) qproj scratch
  unsigned short* attn_bf = (unsigned short*)(ws + qproj_off);
  (void)ws_size; (void)in_sizes; (void)n_in; (void)out_size;

  auto grid1 = [](int n) { return (n + 255) / 256; };

  // 1) casts to bf16 (weights transposed to [N][K])
  cast_f32_bf16<<<grid1(M_ * D_), 256, 0, stream>>>(hidden, hid_bf, M_ * D_);
  cast_transpose_bf16<<<grid1(D_ * H_ * HD_), 256, 0, stream>>>(Wq, wqt, D_, H_ * HD_);
  cast_transpose_bf16<<<grid1(D_ * KVH_ * HD_), 256, 0, stream>>>(Wk, wkt, D_, KVH_ * HD_);
  cast_transpose_bf16<<<grid1(D_ * KVH_ * HD_), 256, 0, stream>>>(Wv, wvt, D_, KVH_ * HD_);
  cast_transpose_bf16<<<grid1(H_ * HD_ * D_), 256, 0, stream>>>(Wo, wot, H_ * HD_, D_);

  // 2) QKV projections (K = 384 for all)
  gemm_bf16_f32<<<dim3(M_ / 64, (H_ * HD_) / 64), 128, 0, stream>>>(hid_bf, wqt, qproj, H_ * HD_, D_);
  gemm_bf16_f32<<<dim3(M_ / 64, (KVH_ * HD_) / 64), 128, 0, stream>>>(hid_bf, wkt, kproj, KVH_ * HD_, D_);
  gemm_bf16_f32<<<dim3(M_ / 64, (KVH_ * HD_) / 64), 128, 0, stream>>>(hid_bf, wvt, vproj, KVH_ * HD_, D_);

  // 3) RoPE + head-major reshape to bf16
  rope_kernel<<<B_ * S_, 128, 0, stream>>>(qproj, kproj, vproj, fc, fs, q_bf, k_bf, v_bf);

  // 4) flash attention
  attn_kernel<<<B_ * H_ * (S_ / 64), 128, 0, stream>>>(q_bf, k_bf, v_bf, attn_bf);

  // 5) output projection -> fp32 d_out
  gemm_bf16_f32<<<dim3(M_ / 64, D_ / 64), 128, 0, stream>>>(attn_bf, wot, (float*)d_out, D_, H_ * HD_);
}
